// EGNN_21973052686852
// MI455X (gfx1250) — compile-verified, matrix operand order checked by reading, not run on hardware
//
#include <hip/hip_runtime.h>
#include <hip/hip_bf16.h>
#include <stdint.h>

// ---- static problem constants ----
#define Bv      64
#define Nv      64
#define Hv      256
#define TEv     128
#define Lv      3
#define SIGMA_  0.5f
#define CR_     (15.0f / 3.0f)   // coords_range / n_layers
#define SMS     264              // LDS row stride in halves (pad vs bank conflicts)

typedef __attribute__((ext_vector_type(16))) _Float16 v16h;
typedef __attribute__((ext_vector_type(8)))  _Float16 v8h;
typedef __attribute__((ext_vector_type(8)))  float    v8f;

__device__ __forceinline__ v8f zero8() {
  v8f z = {0.f,0.f,0.f,0.f,0.f,0.f,0.f,0.f};
  return z;
}
__device__ __forceinline__ float silu_f(float x) { return x / (1.0f + __expf(-x)); }
__device__ __forceinline__ float sigm_f(float x) { return 1.0f / (1.0f + __expf(-x)); }

__device__ __forceinline__ v16h make_frag(const _Float16* p) {
  v8h lo = *(const v8h*)p;
  v8h hi = *(const v8h*)(p + 16);
  v16h r;
#pragma unroll
  for (int i = 0; i < 8; ++i) { r[i] = lo[i]; r[i + 8] = hi[i]; }
  return r;
}

// A fragment (16x32 f16) from a row-major global f16 matrix [M x K]
__device__ __forceinline__ v16h fragA_g(const _Float16* A, int rowBase, int K, int kk, int lane) {
  int row = rowBase + (lane & 15);
  int kb  = kk * 32 + ((lane >> 4) << 3);
  return make_frag(A + (size_t)row * K + kb);
}
// A fragment from LDS tile (row stride SMS halves)
__device__ __forceinline__ v16h fragA_s(const _Float16* sm, int rowBase, int kk, int lane) {
  int row = rowBase + (lane & 15);
  int kb  = kk * 32 + ((lane >> 4) << 3);
  return make_frag(sm + row * SMS + kb);
}
// B fragment from pre-swizzled weights: contiguous 32B per lane
__device__ __forceinline__ v16h fragB(const _Float16* Bsw, int kk, int nt, int ntiles, int lane) {
  return *(const v16h*)(Bsw + (((size_t)(kk * ntiles + nt) * 32) + (size_t)lane) * 16);
}

#define WMMA(a, b, c) \
  __builtin_amdgcn_wmma_f32_16x16x32_f16(false, (a), false, (b), (short)0, (c), false, false)

// ---------------------------------------------------------------------------
// Weight pre-swizzle: f32 row-major [K x 256] -> f16 WMMA-fragment order
// ---------------------------------------------------------------------------
__global__ void k_swizzle(const float* __restrict__ W, _Float16* __restrict__ out,
                          int K, int Nc) {
  int idx = blockIdx.x * blockDim.x + threadIdx.x;
  int ksteps = K >> 5, ntiles = Nc >> 4;
  int total = ksteps * ntiles * 32;
  if (idx >= total) return;
  int lane = idx & 31;
  int t = idx >> 5;
  int nt = t % ntiles;
  int kk = t / ntiles;
  int n  = nt * 16 + (lane & 15);
  int kb = kk * 32 + ((lane >> 4) << 3);
  _Float16* o = out + (size_t)idx * 16;
#pragma unroll
  for (int i = 0; i < 8; ++i) {
    o[i]     = (_Float16)W[(size_t)(kb + i) * Nc + n];
    o[i + 8] = (_Float16)W[(size_t)(kb + 16 + i) * Nc + n];
  }
}

// ---------------------------------------------------------------------------
// Init: sinusoidal time embedding, h0 = temb @ Wemb + bemb (broadcast to the
// 64 nodes of each sample), x0 = xs * c_in
// ---------------------------------------------------------------------------
__global__ void __launch_bounds__(256)
k_init(const float* __restrict__ logt, const float* __restrict__ xs,
       const float* __restrict__ Wemb, const float* __restrict__ bemb,
       float* __restrict__ h, _Float16* __restrict__ h16, float* __restrict__ x0) {
  int b = blockIdx.x, tid = threadIdx.x;
  __shared__ float temb[TEv];
  float lt = logt[b];
  if (tid < TEv) {
    int k = tid & 63;                               // TE/2 = 64 freqs
    float freq = __expf((float)k * (-9.210340371976184f / 64.0f)); // (1e-4)^(k/64)
    float ang  = lt * 0.25f * freq;
    temb[tid]  = (tid < 64) ? __cosf(ang) : __sinf(ang);
  }
  __syncthreads();
  float acc = bemb[tid];
  for (int k = 0; k < TEv; ++k) acc += temb[k] * Wemb[k * Hv + tid];
  _Float16 a16 = (_Float16)acc;
  for (int n = 0; n < Nv; ++n) {
    size_t node = (size_t)b * Nv + n;
    h[node * Hv + tid]   = acc;
    h16[node * Hv + tid] = a16;
  }
  float t   = __expf(lt);
  float cin = 1.0f / sqrtf(SIGMA_ * SIGMA_ + t * t);
  for (int i = tid; i < Nv * 3; i += blockDim.x)
    x0[b * (Nv * 3) + i] = xs[b * (Nv * 3) + i] * cin;
}

// ---------------------------------------------------------------------------
// Node GEMM: C[4096,256](f32) = A[4096,256](f16) @ Bsw   (no bias)
// block = 256 thr = 8 waves; block tile = 64 rows x 256 cols
// wave w: mtile = w&3, ntiles [(w>>2)*8, +8)
// ---------------------------------------------------------------------------
__global__ void __launch_bounds__(256)
k_gemm_p(const _Float16* __restrict__ A, const _Float16* __restrict__ Bsw,
         float* __restrict__ C) {
  int lane = threadIdx.x & 31, wave = threadIdx.x >> 5;
  int mtile = wave & 3, ntb = (wave >> 2) * 8;
  int rowBase = blockIdx.x * 64 + mtile * 16;
  v8f acc[8];
#pragma unroll
  for (int j = 0; j < 8; ++j) acc[j] = zero8();
  for (int kk = 0; kk < 8; ++kk) {
    v16h a = fragA_g(A, rowBase, 256, kk, lane);
#pragma unroll
    for (int j = 0; j < 8; ++j)
      acc[j] = WMMA(a, fragB(Bsw, kk, ntb + j, 16, lane), acc[j]);
  }
  int r0 = rowBase + ((lane >> 4) << 3);
  int c0 = lane & 15;
#pragma unroll
  for (int j = 0; j < 8; ++j) {
    int col = (ntb + j) * 16 + c0;
#pragma unroll
    for (int v = 0; v < 8; ++v)
      C[(size_t)(r0 + v) * Hv + col] = acc[j][v];
  }
}

// Node MLP layer 1: U = silu( h@Wn1a + agg@Wn1b + bn1 )  (K = 256+256)
__global__ void __launch_bounds__(256)
k_gemm_node1(const _Float16* __restrict__ A1, const _Float16* __restrict__ A2,
             const _Float16* __restrict__ B1, const _Float16* __restrict__ B2,
             const float* __restrict__ bias, _Float16* __restrict__ U) {
  int lane = threadIdx.x & 31, wave = threadIdx.x >> 5;
  int mtile = wave & 3, ntb = (wave >> 2) * 8;
  int rowBase = blockIdx.x * 64 + mtile * 16;
  v8f acc[8];
#pragma unroll
  for (int j = 0; j < 8; ++j) acc[j] = zero8();
  for (int kk = 0; kk < 8; ++kk) {
    v16h a = fragA_g(A1, rowBase, 256, kk, lane);
#pragma unroll
    for (int j = 0; j < 8; ++j)
      acc[j] = WMMA(a, fragB(B1, kk, ntb + j, 16, lane), acc[j]);
  }
  for (int kk = 0; kk < 8; ++kk) {
    v16h a = fragA_g(A2, rowBase, 256, kk, lane);
#pragma unroll
    for (int j = 0; j < 8; ++j)
      acc[j] = WMMA(a, fragB(B2, kk, ntb + j, 16, lane), acc[j]);
  }
  int r0 = rowBase + ((lane >> 4) << 3);
  int c0 = lane & 15;
#pragma unroll
  for (int j = 0; j < 8; ++j) {
    int col = (ntb + j) * 16 + c0;
    float bv = bias[col];
#pragma unroll
    for (int v = 0; v < 8; ++v)
      U[(size_t)(r0 + v) * Hv + col] = (_Float16)silu_f(acc[j][v] + bv);
  }
}

// Node MLP layer 2 + residual: h += U@Wn2 + bn2   (updates h f32 and f16)
__global__ void __launch_bounds__(256)
k_gemm_node2(const _Float16* __restrict__ U, const _Float16* __restrict__ Bsw,
             const float* __restrict__ bias,
             float* __restrict__ h, _Float16* __restrict__ h16) {
  int lane = threadIdx.x & 31, wave = threadIdx.x >> 5;
  int mtile = wave & 3, ntb = (wave >> 2) * 8;
  int rowBase = blockIdx.x * 64 + mtile * 16;
  v8f acc[8];
#pragma unroll
  for (int j = 0; j < 8; ++j) acc[j] = zero8();
  for (int kk = 0; kk < 8; ++kk) {
    v16h a = fragA_g(U, rowBase, 256, kk, lane);
#pragma unroll
    for (int j = 0; j < 8; ++j)
      acc[j] = WMMA(a, fragB(Bsw, kk, ntb + j, 16, lane), acc[j]);
  }
  int r0 = rowBase + ((lane >> 4) << 3);
  int c0 = lane & 15;
#pragma unroll
  for (int j = 0; j < 8; ++j) {
    int col = (ntb + j) * 16 + c0;
    float bv = bias[col];
#pragma unroll
    for (int v = 0; v < 8; ++v) {
      size_t idx = (size_t)(r0 + v) * Hv + col;
      float val = acc[j][v] + bv + h[idx];
      h[idx]   = val;
      h16[idx] = (_Float16)val;
    }
  }
}

// ---------------------------------------------------------------------------
// Fused per-node edge pipeline. One block per node (4096 blocks, 256 thr):
//   m1 = silu(P1[r] + P2[c] + radial*wr + be1)      (63 edges + 1 masked pad)
//   m2 = silu(m1 @ We2 + be2)          [WMMA, A in LDS]
//   m  = m2 * sigmoid(m2.Wa + ba)
//   s  = silu(m @ Wc1 + bc1)           [WMMA]
//   trans = diff_n * tanh(s.Wc2) * cr ; x_out[r] = x_in[r] + sum_e trans
//   agg[r] = sum_e m                   (edges per node are contiguous -> no atomics)
// ---------------------------------------------------------------------------
__global__ void __launch_bounds__(256)
k_edge(const float* __restrict__ x_in, const float* __restrict__ P1,
       const float* __restrict__ P2,
       const float* __restrict__ wr,  const float* __restrict__ be1,
       const _Float16* __restrict__ We2sw, const float* __restrict__ be2,
       const float* __restrict__ Wa,  const float* __restrict__ ba,
       const _Float16* __restrict__ Wc1sw, const float* __restrict__ bc1,
       const float* __restrict__ Wc2,
       float* __restrict__ x_out, _Float16* __restrict__ agg16) {
  __shared__ _Float16 sm[64 * SMS];
  __shared__ float rad[64], dnx[64], dny[64], dnz[64], gate[64];
  __shared__ float trx[64], trybuf[64], trz[64];

  int tid  = threadIdx.x;
  int lane = tid & 31, wave = tid >> 5;
  int node = blockIdx.x;
  int b = node >> 6, r = node & 63;

  // 1) edge geometry (edge slot e -> neighbor c; slot 63 is padding)
  if (tid < 64) {
    int e = tid;
    int c = (e >= r) ? e + 1 : e;
    if (c >= 64) c = r;                       // pad slot, masked later
    bool valid = (e < 63);
    float xr0 = x_in[node * 3 + 0], xr1 = x_in[node * 3 + 1], xr2 = x_in[node * 3 + 2];
    int cn = b * 64 + c;
    float d0 = xr0 - x_in[cn * 3 + 0];
    float d1 = xr1 - x_in[cn * 3 + 1];
    float d2 = xr2 - x_in[cn * 3 + 2];
    float radial = d0 * d0 + d1 * d1 + d2 * d2;
    float rinv   = 1.0f / (sqrtf(radial + 1e-8f) + 1.0f);
    if (!valid) { radial = 0.f; rinv = 0.f; }
    rad[e] = radial;
    dnx[e] = d0 * rinv; dny[e] = d1 * rinv; dnz[e] = d2 * rinv;
  }
  __syncthreads();

  // 2) m1 tile into LDS (thread tid owns column tid for all 64 edge rows)
  {
    float p1v = P1[(size_t)node * Hv + tid];
    float wrv = wr[tid];
    float b1  = be1[tid];
    for (int e = 0; e < 64; ++e) {
      int c = (e >= r) ? e + 1 : e;
      if (c >= 64) c = r;
      float p2v = P2[(size_t)(b * 64 + c) * Hv + tid];
      sm[e * SMS + tid] = (_Float16)silu_f(p1v + p2v + rad[e] * wrv + b1);
    }
  }
  __syncthreads();

  int mtile = wave & 3, ntb = (wave >> 2) * 8;
  int r0 = mtile * 16 + ((lane >> 4) << 3);
  int c0 = lane & 15;

  // 3) GEMM1: m2 = silu(m1 @ We2 + be2), written back into sm
  {
    v8f acc[8];
#pragma unroll
    for (int j = 0; j < 8; ++j) acc[j] = zero8();
    for (int kk = 0; kk < 8; ++kk) {
      v16h a = fragA_s(sm, mtile * 16, kk, lane);
#pragma unroll
      for (int j = 0; j < 8; ++j)
        acc[j] = WMMA(a, fragB(We2sw, kk, ntb + j, 16, lane), acc[j]);
    }
    __syncthreads();                          // all A reads done before overwrite
#pragma unroll
    for (int j = 0; j < 8; ++j) {
      int col = (ntb + j) * 16 + c0;
      float bv = be2[col];
#pragma unroll
      for (int v = 0; v < 8; ++v)
        sm[(r0 + v) * SMS + col] = (_Float16)silu_f(acc[j][v] + bv);
    }
  }
  __syncthreads();

  // 4) attention gate and m = m2 * gate
  if (tid < 64) {
    float dot = ba[0];
    for (int j = 0; j < Hv; ++j) dot += (float)sm[tid * SMS + j] * Wa[j];
    gate[tid] = sigm_f(dot);
  }
  __syncthreads();
  for (int e = 0; e < 64; ++e)
    sm[e * SMS + tid] = (_Float16)((float)sm[e * SMS + tid] * gate[e]);
  __syncthreads();

  // 5) agg[node] = sum over real edges of m
  {
    float s = 0.f;
    for (int e = 0; e < 63; ++e) s += (float)sm[e * SMS + tid];
    agg16[(size_t)node * Hv + tid] = (_Float16)s;
  }

  // 6) GEMM2: s = silu(m @ Wc1 + bc1), written back into sm
  {
    v8f acc[8];
#pragma unroll
    for (int j = 0; j < 8; ++j) acc[j] = zero8();
    for (int kk = 0; kk < 8; ++kk) {
      v16h a = fragA_s(sm, mtile * 16, kk, lane);
#pragma unroll
      for (int j = 0; j < 8; ++j)
        acc[j] = WMMA(a, fragB(Wc1sw, kk, ntb + j, 16, lane), acc[j]);
    }
    __syncthreads();
#pragma unroll
    for (int j = 0; j < 8; ++j) {
      int col = (ntb + j) * 16 + c0;
      float bv = bc1[col];
#pragma unroll
      for (int v = 0; v < 8; ++v)
        sm[(r0 + v) * SMS + col] = (_Float16)silu_f(acc[j][v] + bv);
    }
  }
  __syncthreads();

  // 7) tanh-bounded coordinate update; block-local reduce (no atomics)
  if (tid < 64) {
    float dot = 0.f;
    for (int j = 0; j < Hv; ++j) dot += (float)sm[tid * SMS + j] * Wc2[j];
    float th = tanhf(dot) * CR_;
    trx[tid]    = dnx[tid] * th;
    trybuf[tid] = dny[tid] * th;
    trz[tid]    = dnz[tid] * th;
  }
  __syncthreads();
  if (tid < 3) {
    const float* tr = (tid == 0) ? trx : (tid == 1) ? trybuf : trz;
    float s = 0.f;
    for (int e = 0; e < 63; ++e) s += tr[e];
    x_out[node * 3 + tid] = x_in[node * 3 + tid] + s;
  }
}

// ---------------------------------------------------------------------------
// Final EDM recombination with per-sample center-of-gravity removal
// ---------------------------------------------------------------------------
__global__ void __launch_bounds__(192)
k_final(const float* __restrict__ logt, const float* __restrict__ xs,
        const float* __restrict__ xfin, const float* __restrict__ x0,
        float* __restrict__ out) {
  int b = blockIdx.x, tid = threadIdx.x;      // 192 = 64 nodes * 3 dims
  __shared__ float sv[192];
  float lt = logt[b];
  float t  = __expf(lt);
  float denom = SIGMA_ * SIGMA_ + t * t;
  int i = b * 192 + tid;
  float vel = xfin[i] - x0[i];
  sv[tid] = vel;
  __syncthreads();
  int d = tid % 3;
  float mean = 0.f;
  for (int n = 0; n < 64; ++n) mean += sv[n * 3 + d];
  mean *= (1.0f / 64.0f);
  out[i] = xs[i] * (SIGMA_ * SIGMA_ / denom)
         + (vel - mean) * (SIGMA_ * t / sqrtf(denom));
}

// ---------------------------------------------------------------------------
extern "C" void kernel_launch(void* const* d_in, const int* in_sizes, int n_in,
                              void* d_out, int out_size, void* d_ws, size_t ws_size,
                              hipStream_t stream) {
  const float* logt = (const float*)d_in[0];
  const float* xs   = (const float*)d_in[1];
  const float* Wemb = (const float*)d_in[2];
  const float* bemb = (const float*)d_in[3];
  const float* We1  = (const float*)d_in[4];
  const float* be1  = (const float*)d_in[5];
  const float* We2  = (const float*)d_in[6];
  const float* be2  = (const float*)d_in[7];
  const float* Wa   = (const float*)d_in[8];
  const float* ba   = (const float*)d_in[9];
  const float* Wn1  = (const float*)d_in[10];
  const float* bn1  = (const float*)d_in[11];
  const float* Wn2  = (const float*)d_in[12];
  const float* bn2  = (const float*)d_in[13];
  const float* Wc1  = (const float*)d_in[14];
  const float* bc1  = (const float*)d_in[15];
  const float* Wc2  = (const float*)d_in[16];
  float* out = (float*)d_out;
  (void)in_sizes; (void)n_in; (void)out_size; (void)ws_size;

  // bump allocator over d_ws (~21 MB total)
  uintptr_t base = (uintptr_t)d_ws;
  size_t off = 0;
  auto alloc = [&](size_t bytes) -> void* {
    off = (off + 255) & ~(size_t)255;
    void* p = (void*)(base + off);
    off += bytes;
    return p;
  };
  const size_t WSZ = (size_t)256 * 256 * sizeof(_Float16);
  _Float16* sw[Lv][7];
  for (int l = 0; l < Lv; ++l)
    for (int m = 0; m < 7; ++m)
      sw[l][m] = (_Float16*)alloc(WSZ);

  const size_t NN = (size_t)Bv * Nv;              // 4096 nodes
  float*    h     = (float*)   alloc(NN * Hv * 4);
  _Float16* h16   = (_Float16*)alloc(NN * Hv * 2);
  float*    P1    = (float*)   alloc(NN * Hv * 4);
  float*    P2    = (float*)   alloc(NN * Hv * 4);
  _Float16* agg16 = (_Float16*)alloc(NN * Hv * 2);
  _Float16* U16   = (_Float16*)alloc(NN * Hv * 2);
  float*    x0    = (float*)   alloc(NN * 3 * 4);
  float*    xA    = (float*)   alloc(NN * 3 * 4);
  float*    xB    = (float*)   alloc(NN * 3 * 4);

  // pre-swizzle all f16 weight operands into WMMA-fragment order
  for (int l = 0; l < Lv; ++l) {
    const float* We1l = We1 + (size_t)l * 513 * 256;
    const float* Wn1l = Wn1 + (size_t)l * 512 * 256;
    k_swizzle<<<16, 256, 0, stream>>>(We1l,                        sw[l][0], 256, 256);
    k_swizzle<<<16, 256, 0, stream>>>(We1l + 256 * 256,            sw[l][1], 256, 256);
    k_swizzle<<<16, 256, 0, stream>>>(We2 + (size_t)l * 65536,     sw[l][2], 256, 256);
    k_swizzle<<<16, 256, 0, stream>>>(Wc1 + (size_t)l * 65536,     sw[l][3], 256, 256);
    k_swizzle<<<16, 256, 0, stream>>>(Wn1l,                        sw[l][4], 256, 256);
    k_swizzle<<<16, 256, 0, stream>>>(Wn1l + 256 * 256,            sw[l][5], 256, 256);
    k_swizzle<<<16, 256, 0, stream>>>(Wn2 + (size_t)l * 65536,     sw[l][6], 256, 256);
  }

  k_init<<<Bv, 256, 0, stream>>>(logt, xs, Wemb, bemb, h, h16, x0);

  float* xin  = x0;
  float* xout = xA;
  for (int l = 0; l < Lv; ++l) {
    const float* We1l = We1 + (size_t)l * 513 * 256;
    k_gemm_p<<<64, 256, 0, stream>>>(h16, sw[l][0], P1);
    k_gemm_p<<<64, 256, 0, stream>>>(h16, sw[l][1], P2);
    k_edge<<<(int)NN, 256, 0, stream>>>(xin, P1, P2,
        We1l + 512 * 256, be1 + l * 256, sw[l][2], be2 + l * 256,
        Wa + l * 256, ba + l, sw[l][3], bc1 + l * 256, Wc2 + l * 256,
        xout, agg16);
    k_gemm_node1<<<64, 256, 0, stream>>>(h16, agg16, sw[l][4], sw[l][5],
                                         bn1 + l * 256, U16);
    k_gemm_node2<<<64, 256, 0, stream>>>(U16, sw[l][6], bn2 + l * 256, h, h16);
    xin  = xout;
    xout = (l == 0) ? xB : xA;
  }

  k_final<<<Bv, 192, 0, stream>>>(logt, xs, xin, x0, out);
}
